// SuperGlue_44117904064876
// MI455X (gfx1250) — matrix operand order repaired and checked by reading, not compiled, not run on hardware
//
#include <hip/hip_runtime.h>
#include <math.h>

#define D_MODEL 256
#define NHEAD   4
#define DHEAD   64
#define NPTS    1024
#define BATCH   2
#define NP1     1025
#define NLAYERS 18
#define NEG_INF (-__builtin_inff())

typedef __attribute__((ext_vector_type(16))) _Float16 v16h;
typedef __attribute__((ext_vector_type(8)))  float    v8f;

// ---------------------------------------------------------------------------
// Strided WMMA 16x64 strip: 4 accumulators, A fragment loaded once per K-step
// and reused across 4 B subtiles (subtile t shifts B's n-index by 16*t).
// A element (m,k) at A[m*sa_m + k*sa_k]; B element (k,n) at B[k*sb_k + n*sb_n].
// VGPR layouts per CDNA5 ISA 7.12.2 (wave32). BOUND=true adds k-range guards
// (only the K=3 kenc input layer). K-stride==1 operands are fetched as float4
// (b128); those branches are wave-uniform so EXEC stays all-ones for WMMA.
// ---------------------------------------------------------------------------
template <bool BOUND>
__device__ inline void wmma_strip4(const float* __restrict__ A, int sa_m, int sa_k,
                                   const float* __restrict__ B, int sb_k, int sb_n,
                                   int K, v8f acc[4])
{
  const int lane = threadIdx.x & 31;
  const int lo   = lane & 15;
  const int hi   = lane >> 4;
  for (int k0 = 0; k0 < K; k0 += 32) {
    v16h a;
    // ---- A fragment: halves j -> k = k0 + (j<8?0:16) + hi*8 + (j&7)
    if (!BOUND && sa_k == 1 && (sa_m & 3) == 0) {
      const float* Ap = A + lo * sa_m + k0 + hi * 8;
      float4 x0 = *(const float4*)(Ap);
      float4 x1 = *(const float4*)(Ap + 4);
      float4 x2 = *(const float4*)(Ap + 16);
      float4 x3 = *(const float4*)(Ap + 20);
      a[0] = (_Float16)x0.x;  a[1] = (_Float16)x0.y;
      a[2] = (_Float16)x0.z;  a[3] = (_Float16)x0.w;
      a[4] = (_Float16)x1.x;  a[5] = (_Float16)x1.y;
      a[6] = (_Float16)x1.z;  a[7] = (_Float16)x1.w;
      a[8] = (_Float16)x2.x;  a[9] = (_Float16)x2.y;
      a[10] = (_Float16)x2.z; a[11] = (_Float16)x2.w;
      a[12] = (_Float16)x3.x; a[13] = (_Float16)x3.y;
      a[14] = (_Float16)x3.z; a[15] = (_Float16)x3.w;
    } else {
#pragma unroll
      for (int j = 0; j < 16; ++j) {
        const int ka = k0 + ((j < 8) ? 0 : 16) + hi * 8 + (j & 7);
        float av;
        if (BOUND) av = (ka < K) ? A[lo * sa_m + ka * sa_k] : 0.0f;
        else       av = A[lo * sa_m + ka * sa_k];
        a[j] = (_Float16)av;
      }
    }
    // ---- 4 B subtiles: halves j -> k = k0 + (j<8?0:16) + hi*8 + (j&7)
#pragma unroll
    for (int t = 0; t < 4; ++t) {
      const float* Bt = B + t * 16 * sb_n;
      v16h bf;
      if (!BOUND && sb_k == 1 && (sb_n & 3) == 0) {
        const float* Bp = Bt + lo * sb_n + k0 + hi * 8;
        float4 y0 = *(const float4*)(Bp);
        float4 y1 = *(const float4*)(Bp + 4);
        float4 y2 = *(const float4*)(Bp + 16);
        float4 y3 = *(const float4*)(Bp + 20);
        bf[0] = (_Float16)y0.x;  bf[1] = (_Float16)y0.y;
        bf[2] = (_Float16)y0.z;  bf[3] = (_Float16)y0.w;
        bf[4] = (_Float16)y1.x;  bf[5] = (_Float16)y1.y;
        bf[6] = (_Float16)y1.z;  bf[7] = (_Float16)y1.w;
        bf[8] = (_Float16)y2.x;  bf[9] = (_Float16)y2.y;
        bf[10] = (_Float16)y2.z; bf[11] = (_Float16)y2.w;
        bf[12] = (_Float16)y3.x; bf[13] = (_Float16)y3.y;
        bf[14] = (_Float16)y3.z; bf[15] = (_Float16)y3.w;
      } else {
#pragma unroll
        for (int j = 0; j < 16; ++j) {
          const int kb = k0 + ((j < 8) ? 0 : 16) + hi * 8 + (j & 7);
          float bv;
          if (BOUND) bv = (kb < K) ? Bt[kb * sb_k + lo * sb_n] : 0.0f;
          else       bv = Bt[kb * sb_k + lo * sb_n];
          bf[j] = (_Float16)bv;
        }
      }
      acc[t] = __builtin_amdgcn_wmma_f32_16x16x32_f16(false, a, false, bf,
                                                      (short)0, acc[t], false, false);
    }
  }
}

// Y[b,o,n] = sum_i W[o,i]*X[b,i,n] + bias[o]   (optionally +=)
// block (32,4): 4 waves, each owns a 16(o) x 64(n) strip
template <bool BOUND>
__global__ void k_conv1(const float* __restrict__ W, const float* __restrict__ bias,
                        const float* __restrict__ X, float* __restrict__ Y,
                        int O, int I, int N, int accum)
{
  const int n0 = (blockIdx.x * blockDim.y + threadIdx.y) * 64;
  const int o0 = blockIdx.y * 16;
  const int b  = blockIdx.z;
  v8f acc[4] = {};
  wmma_strip4<BOUND>(W + o0 * I, I, 1, X + (size_t)b * I * N + n0, N, 1, I, acc);
  const int lane = threadIdx.x & 31, lo = lane & 15, hi = lane >> 4;
  float* Yb = Y + (size_t)b * O * N;
#pragma unroll
  for (int t = 0; t < 4; ++t)
#pragma unroll
    for (int r = 0; r < 8; ++r) {
      int o = o0 + r + hi * 8, n = n0 + t * 16 + lo;
      float val = acc[t][r] + bias[o];
      if (accum) Yb[o * N + n] += val; else Yb[o * N + n] = val;
    }
}

// S[bhz,n,m] = (Q_h^T K_h)/8 masked; q,k layout (B,D,N), head h uses d = dh*H+h
// wave strip: 16(n) x 64(m)
__global__ void k_attn_scores(const float* __restrict__ Q, const float* __restrict__ Kx,
                              const float* __restrict__ mx, const float* __restrict__ ms,
                              float* __restrict__ S, int bh0)
{
  const int n0 = (blockIdx.x * blockDim.y + threadIdx.y) * 16;
  const int m0 = blockIdx.y * 64;
  const int bh = bh0 + blockIdx.z;
  const int b = bh / NHEAD, h = bh % NHEAD;
  v8f acc[4] = {};
  wmma_strip4<false>(Q + (size_t)b * D_MODEL * NPTS + h * NPTS + n0, 1, NHEAD * NPTS,
                     Kx + (size_t)b * D_MODEL * NPTS + h * NPTS + m0, NHEAD * NPTS, 1,
                     DHEAD, acc);
  const int lane = threadIdx.x & 31, lo = lane & 15, hi = lane >> 4;
  float* Sb = S + (size_t)blockIdx.z * NPTS * NPTS;
  float mxv[8];
#pragma unroll
  for (int r = 0; r < 8; ++r) mxv[r] = mx[b * NPTS + n0 + r + hi * 8];
#pragma unroll
  for (int t = 0; t < 4; ++t) {
    int m = m0 + t * 16 + lo;
    float msv = ms[b * NPTS + m];
#pragma unroll
    for (int r = 0; r < 8; ++r) {
      int n = n0 + r + hi * 8;
      float val = (mxv[r] * msv > 0.0f) ? acc[t][r] * 0.125f : NEG_INF;
      Sb[(size_t)n * NPTS + m] = val;
    }
  }
}

// row softmax in-place; fully-masked (-inf) rows -> zeros (NaN->0 rule)
__global__ void k_softmax(float* __restrict__ S, int rowlen)
{
  float* p = S + (size_t)blockIdx.x * rowlen;
  __shared__ float red[256];
  float m = NEG_INF;
  for (int j = threadIdx.x; j < rowlen; j += blockDim.x) m = fmaxf(m, p[j]);
  red[threadIdx.x] = m; __syncthreads();
  for (int s = blockDim.x / 2; s > 0; s >>= 1) {
    if (threadIdx.x < s) red[threadIdx.x] = fmaxf(red[threadIdx.x], red[threadIdx.x + s]);
    __syncthreads();
  }
  m = red[0]; __syncthreads();
  float sum = 0.f;
  if (m > NEG_INF)
    for (int j = threadIdx.x; j < rowlen; j += blockDim.x) sum += __expf(p[j] - m);
  red[threadIdx.x] = sum; __syncthreads();
  for (int s = blockDim.x / 2; s > 0; s >>= 1) {
    if (threadIdx.x < s) red[threadIdx.x] += red[threadIdx.x + s];
    __syncthreads();
  }
  sum = red[0];
  float inv = (m > NEG_INF && sum > 0.f) ? 1.0f / sum : 0.0f;
  for (int j = threadIdx.x; j < rowlen; j += blockDim.x) {
    float e = (m > NEG_INF) ? __expf(p[j] - m) : 0.0f;
    p[j] = e * inv;
  }
}

// O[b, dh*H+h, n] = sum_m P[n,m] * V[b, dh*H+h, m]; wave strip 16(dh) x 64(n)
__global__ void k_attn_out(const float* __restrict__ V, const float* __restrict__ S,
                           float* __restrict__ O, int bh0)
{
  const int n0 = (blockIdx.x * blockDim.y + threadIdx.y) * 64;
  const int dh0 = blockIdx.y * 16;
  const int bh = bh0 + blockIdx.z;
  const int b = bh / NHEAD, h = bh % NHEAD;
  const float* Sb = S + (size_t)blockIdx.z * NPTS * NPTS;
  v8f acc[4] = {};
  wmma_strip4<false>(V + (size_t)b * D_MODEL * NPTS + (dh0 * NHEAD + h) * NPTS, NHEAD * NPTS, 1,
                     Sb + (size_t)n0 * NPTS, 1, NPTS,
                     NPTS, acc);
  const int lane = threadIdx.x & 31, lo = lane & 15, hi = lane >> 4;
#pragma unroll
  for (int t = 0; t < 4; ++t)
#pragma unroll
    for (int r = 0; r < 8; ++r) {
      int dh = dh0 + r + hi * 8, n = n0 + t * 16 + lo;
      O[(size_t)b * D_MODEL * NPTS + (dh * NHEAD + h) * NPTS + n] = acc[t][r];
    }
}

__global__ void k_bn_stats(const float* __restrict__ X, const float* __restrict__ mask,
                           float* __restrict__ mean, float* __restrict__ var, int C, int N)
{
  const int c = blockIdx.x;
  __shared__ float r1[256], r2[256], r3[256];
  float s1 = 0, s2 = 0, cnt = 0;
  for (int t = threadIdx.x; t < BATCH * N; t += blockDim.x) {
    int b = t / N, n = t % N;
    float mk = mask[b * N + n];
    float x = X[((size_t)b * C + c) * N + n];
    s1 += x * mk; s2 += x * x * mk; cnt += mk;
  }
  r1[threadIdx.x] = s1; r2[threadIdx.x] = s2; r3[threadIdx.x] = cnt; __syncthreads();
  for (int s = blockDim.x / 2; s > 0; s >>= 1) {
    if (threadIdx.x < s) {
      r1[threadIdx.x] += r1[threadIdx.x + s];
      r2[threadIdx.x] += r2[threadIdx.x + s];
      r3[threadIdx.x] += r3[threadIdx.x + s];
    }
    __syncthreads();
  }
  if (threadIdx.x == 0) {
    float mu = r1[0] / r3[0];
    mean[c] = mu;
    var[c]  = r2[0] / r3[0] - mu * mu;
  }
}

__global__ void k_bn_relu(float* __restrict__ X, const float* __restrict__ mean,
                          const float* __restrict__ var, const float* __restrict__ g,
                          const float* __restrict__ bt, int C, int N, long total)
{
  long i = (long)blockIdx.x * blockDim.x + threadIdx.x;
  if (i >= total) return;
  int c = (int)((i / N) % C);
  float xh = (X[i] - mean[c]) * rsqrtf(var[c] + 1e-5f);
  X[i] = fmaxf(g[c] * xh + bt[c], 0.0f);
}

__global__ void k_kenc_input(const float* __restrict__ kpts, const float* __restrict__ sc,
                             const int* __restrict__ Hp, const int* __restrict__ Wp,
                             float* __restrict__ X)
{
  int i = blockIdx.x * blockDim.x + threadIdx.x;
  if (i >= BATCH * NPTS) return;
  int b = i / NPTS, n = i % NPTS;
  float w = (float)Wp[0], h = (float)Hp[0];
  float scl = fmaxf(w, h) * 0.7f;
  float kx = kpts[(size_t)(b * NPTS + n) * 2 + 0];
  float ky = kpts[(size_t)(b * NPTS + n) * 2 + 1];
  X[((size_t)b * 3 + 0) * NPTS + n] = (kx - w * 0.5f) / scl;
  X[((size_t)b * 3 + 1) * NPTS + n] = (ky - h * 0.5f) / scl;
  X[((size_t)b * 3 + 2) * NPTS + n] = sc[b * NPTS + n];
}

__global__ void k_add2(const float* __restrict__ a, const float* __restrict__ b,
                       float* __restrict__ y, long n)
{
  long i = (long)blockIdx.x * blockDim.x + threadIdx.x;
  if (i < n) y[i] = a[i] + b[i];
}

__global__ void k_addto(float* __restrict__ y, const float* __restrict__ x, long n)
{
  long i = (long)blockIdx.x * blockDim.x + threadIdx.x;
  if (i < n) y[i] += x[i];
}

__global__ void k_zero(float* __restrict__ p, long n)
{
  long i = (long)blockIdx.x * blockDim.x + threadIdx.x;
  if (i < n) p[i] = 0.0f;
}

__global__ void k_concat(const float* __restrict__ x, const float* __restrict__ m,
                         float* __restrict__ h, long total)
{
  long i = (long)blockIdx.x * blockDim.x + threadIdx.x;
  if (i >= total) return;
  int n = (int)(i % NPTS);
  int c = (int)((i / NPTS) % (2 * D_MODEL));
  int b = (int)(i / ((long)NPTS * 2 * D_MODEL));
  float v = (c < D_MODEL) ? x[((size_t)b * D_MODEL + c) * NPTS + n]
                          : m[((size_t)b * D_MODEL + (c - D_MODEL)) * NPTS + n];
  h[i] = v;
}

// raw match scores into coupling matrix interior: Z[b,n,m] = md0.md1 / sqrt(D)
__global__ void k_final_scores(const float* __restrict__ md0, const float* __restrict__ md1,
                               float* __restrict__ Z)
{
  const int n0 = (blockIdx.x * blockDim.y + threadIdx.y) * 16;
  const int m0 = blockIdx.y * 64;
  const int b = blockIdx.z;
  v8f acc[4] = {};
  wmma_strip4<false>(md0 + (size_t)b * D_MODEL * NPTS + n0, 1, NPTS,
                     md1 + (size_t)b * D_MODEL * NPTS + m0, NPTS, 1,
                     D_MODEL, acc);
  const int lane = threadIdx.x & 31, lo = lane & 15, hi = lane >> 4;
  float* Zb = Z + (size_t)b * NP1 * NP1;
#pragma unroll
  for (int t = 0; t < 4; ++t)
#pragma unroll
    for (int r = 0; r < 8; ++r) {
      int n = n0 + r + hi * 8, m = m0 + t * 16 + lo;
      Zb[(size_t)n * NP1 + m] = acc[t][r] * 0.0625f; // 1/sqrt(256)
    }
}

__global__ void k_couple(float* __restrict__ Z, const float* __restrict__ m0,
                         const float* __restrict__ m1, const float* __restrict__ alpha)
{
  long i = (long)blockIdx.x * blockDim.x + threadIdx.x;
  if (i >= (long)BATCH * NP1 * NP1) return;
  int j = (int)(i % NP1);
  int r = (int)((i / NP1) % NP1);
  int b = (int)(i / ((long)NP1 * NP1));
  float a = alpha[0], v;
  if (r < NPTS && j < NPTS)      v = Z[i] + logf(m0[b * NPTS + r] * m1[b * NPTS + j] + 1e-45f);
  else if (r < NPTS)             v = (m0[b * NPTS + r] > 0.f) ? a : NEG_INF;
  else if (j < NPTS)             v = (m1[b * NPTS + j] > 0.f) ? a : NEG_INF;
  else                           v = a;
  Z[i] = v;
}

// LDS-tiled transpose: Zt[b,j,i] = Z[b,i,j]  (so both sinkhorn passes are row LSEs)
__global__ void k_transpose(const float* __restrict__ Z, float* __restrict__ Zt)
{
  __shared__ float tile[32][33];
  const int b = blockIdx.z;
  const int i0 = blockIdx.y * 32, j0 = blockIdx.x * 32;
  const float* Zb = Z + (size_t)b * NP1 * NP1;
  float* Ztb = Zt + (size_t)b * NP1 * NP1;
  for (int r = threadIdx.y; r < 32; r += 8) {
    int i = i0 + r, j = j0 + threadIdx.x;
    if (i < NP1 && j < NP1) tile[r][threadIdx.x] = Zb[(size_t)i * NP1 + j];
  }
  __syncthreads();
  for (int r = threadIdx.y; r < 32; r += 8) {
    int j = j0 + r, i = i0 + threadIdx.x;
    if (i < NP1 && j < NP1) Ztb[(size_t)j * NP1 + i] = tile[threadIdx.x][r];
  }
}

__global__ void k_prep_mu(const float* __restrict__ m0, const float* __restrict__ m1,
                          float* __restrict__ logmu, float* __restrict__ lognu,
                          float* __restrict__ normv)
{
  int b = blockIdx.x;
  __shared__ float r0[256], r1[256];
  float s0 = 0, s1 = 0;
  for (int n = threadIdx.x; n < NPTS; n += blockDim.x) {
    s0 += m0[b * NPTS + n]; s1 += m1[b * NPTS + n];
  }
  r0[threadIdx.x] = s0; r1[threadIdx.x] = s1; __syncthreads();
  for (int s = blockDim.x / 2; s > 0; s >>= 1) {
    if (threadIdx.x < s) { r0[threadIdx.x] += r0[threadIdx.x + s]; r1[threadIdx.x] += r1[threadIdx.x + s]; }
    __syncthreads();
  }
  float ms = r0[0], ns = r1[0];
  float norm = -logf(ms + ns);
  for (int n = threadIdx.x; n < NPTS; n += blockDim.x) {
    logmu[b * NP1 + n] = (m0[b * NPTS + n] > 0.f) ? norm : NEG_INF;
    lognu[b * NP1 + n] = (m1[b * NPTS + n] > 0.f) ? norm : NEG_INF;
  }
  if (threadIdx.x == 0) {
    logmu[b * NP1 + NPTS] = logf(ns) + norm;
    lognu[b * NP1 + NPTS] = logf(ms) + norm;
    normv[b] = norm;
  }
}

// out[b,r] = logT[b,r] - lse_j(M[b,r,j] + dual[b,j]); non-finite -> -inf
// Used for BOTH sinkhorn half-steps (row-major M; pass Zt for the column step).
__global__ void k_sink_row(const float* __restrict__ M, const float* __restrict__ dual,
                           const float* __restrict__ logT, float* __restrict__ out)
{
  int b = blockIdx.x / NP1, r = blockIdx.x % NP1;
  const float* Mr = M + ((size_t)b * NP1 + r) * NP1;
  const float* db = dual + b * NP1;
  __shared__ float red[256];
  float m = NEG_INF;
  for (int j = threadIdx.x; j < NP1; j += blockDim.x) m = fmaxf(m, Mr[j] + db[j]);
  red[threadIdx.x] = m; __syncthreads();
  for (int s = blockDim.x / 2; s > 0; s >>= 1) {
    if (threadIdx.x < s) red[threadIdx.x] = fmaxf(red[threadIdx.x], red[threadIdx.x + s]);
    __syncthreads();
  }
  m = red[0]; __syncthreads();
  float sum = 0.f;
  if (m > NEG_INF)
    for (int j = threadIdx.x; j < NP1; j += blockDim.x) sum += __expf(Mr[j] + db[j] - m);
  red[threadIdx.x] = sum; __syncthreads();
  for (int s = blockDim.x / 2; s > 0; s >>= 1) {
    if (threadIdx.x < s) red[threadIdx.x] += red[threadIdx.x + s];
    __syncthreads();
  }
  if (threadIdx.x == 0) {
    float lse = (m > NEG_INF) ? (m + logf(red[0])) : NEG_INF;
    float res = logT[b * NP1 + r] - lse;
    out[b * NP1 + r] = __builtin_isfinite(res) ? res : NEG_INF;
  }
}

__global__ void k_output(const float* __restrict__ Z, const float* __restrict__ uu,
                         const float* __restrict__ vv, const float* __restrict__ normv,
                         float* __restrict__ out)
{
  long i = (long)blockIdx.x * blockDim.x + threadIdx.x;
  if (i >= (long)BATCH * NP1 * NP1) return;
  int j = (int)(i % NP1);
  int r = (int)((i / NP1) % NP1);
  int b = (int)(i / ((long)NP1 * NP1));
  out[i] = Z[i] + uu[b * NP1 + r] + vv[b * NP1 + j] - normv[b];
}

// ---------------------------------------------------------------------------
// Host side
// ---------------------------------------------------------------------------
struct GnnP { const float *pw, *pb, *mw, *mb, *w1, *b1, *w2, *b2, *g, *beta; };
struct AllP {
  const float *bin, *fw, *fb;
  const float *kw[4], *kbias[4], *kg[3], *kbeta[3];
  GnnP l[NLAYERS];
};

static void build_params(void* const* d_in, const int* in_sizes, int n_in, AllP& P)
{
  auto f = [&](int i) { return (const float*)d_in[i]; };
  int i = 10;
  int tag = (n_in > 10) ? in_sizes[10] : 1;
  if (tag == 96) { // full insertion order
    for (int k = 0; k < 4; ++k) P.kw[k] = f(i++);
    for (int k = 0; k < 4; ++k) P.kbias[k] = f(i++);
    for (int k = 0; k < 3; ++k) P.kg[k] = f(i++);
    for (int k = 0; k < 3; ++k) P.kbeta[k] = f(i++);
    for (int L = 0; L < NLAYERS; ++L) {
      GnnP& g = P.l[L];
      g.pw = f(i++); g.pb = f(i++); g.mw = f(i++); g.mb = f(i++);
      g.w1 = f(i++); g.b1 = f(i++); g.g = f(i++); g.beta = f(i++);
      g.w2 = f(i++); g.b2 = f(i++);
    }
    P.fw = f(i++); P.fb = f(i++); P.bin = f(i++);
  } else if (tag == 32) { // insertion outer, sorted inner dicts
    for (int k = 0; k < 4; ++k) P.kbias[k] = f(i++);
    for (int k = 0; k < 3; ++k) P.kbeta[k] = f(i++);
    for (int k = 0; k < 3; ++k) P.kg[k] = f(i++);
    for (int k = 0; k < 4; ++k) P.kw[k] = f(i++);
    for (int L = 0; L < NLAYERS; ++L) {
      GnnP& g = P.l[L];
      g.beta = f(i++); g.g = f(i++); g.mb = f(i++); g.mw = f(i++);
      g.b1 = f(i++); g.b2 = f(i++); g.w1 = f(i++); g.w2 = f(i++);
      g.pb = f(i++); g.pw = f(i++);
    }
    P.fw = f(i++); P.fb = f(i++); P.bin = f(i++);
  } else { // tag == 1: fully sorted jax tree order (default)
    P.bin = f(i++); P.fb = f(i++); P.fw = f(i++);
    for (int L = 0; L < NLAYERS; ++L) {
      GnnP& g = P.l[L];
      g.beta = f(i++); g.g = f(i++); g.mb = f(i++); g.mw = f(i++);
      g.b1 = f(i++); g.b2 = f(i++); g.w1 = f(i++); g.w2 = f(i++);
      g.pb = f(i++); g.pw = f(i++);
    }
    for (int k = 0; k < 4; ++k) P.kbias[k] = f(i++);
    for (int k = 0; k < 3; ++k) P.kbeta[k] = f(i++);
    for (int k = 0; k < 3; ++k) P.kg[k] = f(i++);
    for (int k = 0; k < 4; ++k) P.kw[k] = f(i++);
  }
}

extern "C" void kernel_launch(void* const* d_in, const int* in_sizes, int n_in,
                              void* d_out, int out_size, void* d_ws, size_t ws_size,
                              hipStream_t stream)
{
  (void)out_size;
  const float* desc0 = (const float*)d_in[0];
  const float* desc1 = (const float*)d_in[1];
  const float* kpts0 = (const float*)d_in[2];
  const float* kpts1 = (const float*)d_in[3];
  const float* scr0  = (const float*)d_in[4];
  const float* scr1  = (const float*)d_in[5];
  const float* mask0 = (const float*)d_in[6];
  const float* mask1 = (const float*)d_in[7];
  const int*   hh    = (const int*)d_in[8];
  const int*   ww    = (const int*)d_in[9];

  AllP pp;
  build_params(d_in, in_sizes, n_in, pp);

  // workspace carve (floats)
  const size_t BDN = (size_t)BATCH * D_MODEL * NPTS;
  float* base = (float*)d_ws;
  size_t off = 0;
  auto carve = [&](size_t n) { float* p = base + off; off += n; return p; };
  float* d0w = carve(BDN);  float* d1w = carve(BDN);
  float* qb  = carve(BDN);  float* kb  = carve(BDN);  float* vb = carve(BDN);
  float* ob  = carve(BDN);  float* msg = carve(BDN);
  float* hb  = carve(2 * BDN); float* tb = carve(2 * BDN);
  float* dl0 = carve(BDN);  float* dl1 = carve(BDN);
  float* md0 = carve(BDN);  float* md1 = carve(BDN);
  float* meanb = carve(512); float* varb = carve(512);
  float* logmu = carve(BATCH * NP1); float* lognu = carve(BATCH * NP1);
  float* uu = carve(BATCH * NP1);    float* vvv = carve(BATCH * NP1);
  float* normv = carve(8);
  float* Zb = carve((size_t)BATCH * NP1 * NP1);
  float* Zt = carve((size_t)BATCH * NP1 * NP1);
  size_t avail = (ws_size / 4 > off) ? (ws_size / 4 - off) : 0;
  const size_t SLAB = (size_t)NPTS * NPTS;
  int nbh = (avail >= 8 * SLAB) ? 8 : 1;
  float* Sb = carve((size_t)nbh * SLAB);

  const int TPB = 256;
  auto cdiv = [](long a, long b) { return (int)((a + b - 1) / b); };
  const dim3 wblk(32, 4);                 // 4 waves/block, one 16x64 strip each
  dim3 gD(NPTS / 256, D_MODEL / 16, BATCH);

  // ---------------- keypoint encoder ----------------
  auto kenc = [&](const float* kpts, const float* scr, const float* mask,
                  const float* desc, float* dout) {
    k_kenc_input<<<cdiv(BATCH * NPTS, TPB), TPB, 0, stream>>>(kpts, scr, hh, ww, qb);
    const int ch[5] = {3, 32, 64, 128, 256};
    float* bufs[2] = {qb, kb};
    int cur = 0;
    for (int li = 0; li < 4; ++li) {
      float* src = bufs[cur]; float* dst = bufs[1 - cur];
      dim3 g(NPTS / 256, ch[li + 1] / 16, BATCH);
      if (li == 0)
        k_conv1<true><<<g, wblk, 0, stream>>>(pp.kw[li], pp.kbias[li], src, dst,
                                              ch[li + 1], ch[li], NPTS, 0);
      else
        k_conv1<false><<<g, wblk, 0, stream>>>(pp.kw[li], pp.kbias[li], src, dst,
                                               ch[li + 1], ch[li], NPTS, 0);
      if (li < 3) {
        k_bn_stats<<<ch[li + 1], 256, 0, stream>>>(dst, mask, meanb, varb, ch[li + 1], NPTS);
        long tot = (long)BATCH * ch[li + 1] * NPTS;
        k_bn_relu<<<cdiv(tot, TPB), TPB, 0, stream>>>(dst, meanb, varb, pp.kg[li],
                                                      pp.kbeta[li], ch[li + 1], NPTS, tot);
      }
      cur = 1 - cur;
    }
    k_add2<<<cdiv((long)BDN, TPB), TPB, 0, stream>>>(desc, bufs[cur], dout, (long)BDN);
  };
  kenc(kpts0, scr0, mask0, desc0, d0w);
  kenc(kpts1, scr1, mask1, desc1, d1w);

  // ---------------- GNN layers ----------------
  auto prop = [&](const float* x, const float* src, const float* xm, const float* sm,
                  const GnnP& lp, float* delta) {
    k_conv1<false><<<gD, wblk, 0, stream>>>(lp.pw,                         lp.pb,       x,   qb, 256, 256, NPTS, 0);
    k_conv1<false><<<gD, wblk, 0, stream>>>(lp.pw + D_MODEL * D_MODEL,     lp.pb + 256, src, kb, 256, 256, NPTS, 0);
    k_conv1<false><<<gD, wblk, 0, stream>>>(lp.pw + 2 * D_MODEL * D_MODEL, lp.pb + 512, src, vb, 256, 256, NPTS, 0);
    for (int bh0 = 0; bh0 < BATCH * NHEAD; bh0 += nbh) {
      dim3 gs(NPTS / 64, NPTS / 64, nbh);
      k_attn_scores<<<gs, wblk, 0, stream>>>(qb, kb, xm, sm, Sb, bh0);
      k_softmax<<<nbh * NPTS, 256, 0, stream>>>(Sb, NPTS);
      dim3 go(NPTS / 256, DHEAD / 16, nbh);
      k_attn_out<<<go, wblk, 0, stream>>>(vb, Sb, ob, bh0);
    }
    k_conv1<false><<<gD, wblk, 0, stream>>>(lp.mw, lp.mb, ob, msg, 256, 256, NPTS, 0);
    long totH = (long)BATCH * 2 * D_MODEL * NPTS;
    k_concat<<<cdiv(totH, TPB), TPB, 0, stream>>>(x, msg, hb, totH);
    dim3 gH(NPTS / 256, 512 / 16, BATCH);
    k_conv1<false><<<gH, wblk, 0, stream>>>(lp.w1, lp.b1, hb, tb, 512, 512, NPTS, 0);
    k_bn_stats<<<512, 256, 0, stream>>>(tb, xm, meanb, varb, 512, NPTS);
    k_bn_relu<<<cdiv(totH, TPB), TPB, 0, stream>>>(tb, meanb, varb, lp.g, lp.beta, 512, NPTS, totH);
    k_conv1<false><<<gD, wblk, 0, stream>>>(lp.w2, lp.b2, tb, delta, 256, 512, NPTS, 0);
  };

  for (int li = 0; li < NLAYERS; ++li) {
    bool cross = (li & 1) != 0; // NAMES = ['self','cross']*9
    const GnnP& lp = pp.l[li];
    prop(d0w, cross ? d1w : d0w, mask0, cross ? mask1 : mask0, lp, dl0);
    prop(d1w, cross ? d0w : d1w, mask1, cross ? mask0 : mask1, lp, dl1);
    k_addto<<<cdiv((long)BDN, TPB), TPB, 0, stream>>>(d0w, dl0, (long)BDN);
    k_addto<<<cdiv((long)BDN, TPB), TPB, 0, stream>>>(d1w, dl1, (long)BDN);
  }

  // ---------------- final projection + optimal transport ----------------
  k_conv1<false><<<gD, wblk, 0, stream>>>(pp.fw, pp.fb, d0w, md0, 256, 256, NPTS, 0);
  k_conv1<false><<<gD, wblk, 0, stream>>>(pp.fw, pp.fb, d1w, md1, 256, 256, NPTS, 0);
  dim3 gz(NPTS / 64, NPTS / 64, BATCH);
  k_final_scores<<<gz, wblk, 0, stream>>>(md0, md1, Zb);
  long totZ = (long)BATCH * NP1 * NP1;
  k_couple<<<cdiv(totZ, TPB), TPB, 0, stream>>>(Zb, mask0, mask1, pp.bin);
  dim3 gt(cdiv(NP1, 32), cdiv(NP1, 32), BATCH);
  k_transpose<<<gt, dim3(32, 8), 0, stream>>>(Zb, Zt);
  k_prep_mu<<<BATCH, 256, 0, stream>>>(mask0, mask1, logmu, lognu, normv);
  k_zero<<<cdiv((long)BATCH * NP1, TPB), TPB, 0, stream>>>(uu, BATCH * NP1);
  k_zero<<<cdiv((long)BATCH * NP1, TPB), TPB, 0, stream>>>(vvv, BATCH * NP1);
  for (int it = 0; it < 100; ++it) {
    k_sink_row<<<BATCH * NP1, 256, 0, stream>>>(Zb, vvv, logmu, uu);   // update u
    k_sink_row<<<BATCH * NP1, 256, 0, stream>>>(Zt, uu, lognu, vvv);   // update v
  }
  k_output<<<cdiv(totZ, TPB), TPB, 0, stream>>>(Zb, uu, vvv, normv, (float*)d_out);
}
